// ExphormerAttention_51007031607316
// MI455X (gfx1250) — matrix operand order repaired and checked by reading, not compiled, not run on hardware
//
#include <hip/hip_runtime.h>
#include <hip/hip_bf16.h>

// ---------------------------------------------------------------------------
// Exphormer graph attention for MI455X (gfx1250, wave32, WMMA).
//   N=100k nodes, NE=1.6M edges, IN=OUT=64, H=8, D=8.
//   * Node projections Q/K/V: bf16 WMMA 16x16x32 GEMM -> workspace (L2-resident).
//   * Edge kernel: fuse E = edge_attr@WE+bE via WMMA per 16-edge tile (saves
//     ~820MB HBM), stage in LDS, then gather/score/exp/scatter with f32 atomics.
//   * Persistent grid-stride waves: weight B-fragments + biases live in
//     registers across tiles; only wave-local s_wait_dscnt for LDS ordering.
// ---------------------------------------------------------------------------

#define DIM 64          // IN_DIM == OUT_DIM
#define HDIM 8          // head dim
#define ELDS_STRIDE 66  // pad 64->66 floats to spread LDS banks

typedef __bf16 bf16_t;
typedef bf16_t v16bf __attribute__((ext_vector_type(16)));
typedef float  v8f   __attribute__((ext_vector_type(8)));

// A fragment: 16x32 bf16 (rows x K), fp32 row-major source, cvt in-register.
// Lane l: m=l&15, hi=l>>4; elems 0..7 -> K=kbase+8*hi+i, 8..15 -> K=kbase+16+8*hi+i.
__device__ __forceinline__ v16bf load_a_bf16(const float* __restrict__ src, int ld,
                                             int row0, int nrows, int kbase, int lane) {
  int m = row0 + (lane & 15);
  if (m > nrows - 1) m = nrows - 1;   // clamp (tails only)
  int hi = lane >> 4;
  const float* p = src + (long long)m * ld;
  v16bf a;
#pragma unroll
  for (int i = 0; i < 8; ++i) a[i] = (bf16_t)p[kbase + 8 * hi + i];
#pragma unroll
  for (int i = 0; i < 8; ++i) a[8 + i] = (bf16_t)p[kbase + 16 + 8 * hi + i];
  return a;
}

// B fragment: 32x16 bf16 of row-major W[64][64]; lane col fixed, K = 8*hi + e (e<8), 16 + 8*hi + (e-8) (e>=8).
__device__ __forceinline__ v16bf load_b_bf16(const float* __restrict__ W, int ldw,
                                             int kbase, int col, int lane) {
  int hi = lane >> 4;
  v16bf b;
#pragma unroll
  for (int e = 0; e < 16; ++e)
    b[e] = (bf16_t)W[(long long)(kbase + 8 * hi + e + (e >= 8 ? 8 : 0)) * ldw + col];
  return b;
}

typedef float v4f __attribute__((ext_vector_type(4)));

__global__ void zero_f32_kernel(float* __restrict__ p, long long n4) {
  long long i = (long long)blockIdx.x * blockDim.x + threadIdx.x;
  if (i < n4) ((v4f*)p)[i] = (v4f){0.f, 0.f, 0.f, 0.f};
}

// ---------------------------------------------------------------------------
// Node projections: Q/K/V = x @ W{Q,K,V} + b.
// 12 waves/block: wave = 4*mat + colTile; persistent over row tiles so the
// B-fragments (weights) and bias stay in registers.
// ---------------------------------------------------------------------------
__global__ void __launch_bounds__(384)
node_proj_kernel(const float* __restrict__ x,
                 const float* __restrict__ Wq, const float* __restrict__ bq,
                 const float* __restrict__ Wk, const float* __restrict__ bk,
                 const float* __restrict__ Wv, const float* __restrict__ bv,
                 float* __restrict__ Qd, float* __restrict__ Kd, float* __restrict__ Vd,
                 int n, int ntiles) {
  const int lane = threadIdx.x & 31;
  const int wave = threadIdx.x >> 5;   // 0..11
  const int mat  = wave >> 2;          // 0=Q 1=K 2=V
  const int ct   = wave & 3;           // output column tile (16 cols)
  const int nl   = lane & 15;
  const int hi   = lane >> 4;

  const float* W    = (mat == 0) ? Wq : (mat == 1) ? Wk : Wv;
  const float* bias = (mat == 0) ? bq : (mat == 1) ? bk : bv;
  float*       out  = (mat == 0) ? Qd : (mat == 1) ? Kd : Vd;

  const int col = ct * 16 + nl;
  // loop-invariant weight fragments + bias (registers)
  v16bf bw0 = load_b_bf16(W, DIM, 0, col, lane);
  v16bf bw1 = load_b_bf16(W, DIM, 32, col, lane);
  const float bb = bias[col];

  for (int tile = blockIdx.x; tile < ntiles; tile += gridDim.x) {
    const int row0 = tile * 16;
    v8f c = {};
    {
      v16bf a = load_a_bf16(x, DIM, row0, n, 0, lane);
      c = __builtin_amdgcn_wmma_f32_16x16x32_bf16(false, a, false, bw0,
                                                  (short)0, c, false, false);
    }
    {
      v16bf a = load_a_bf16(x, DIM, row0, n, 32, lane);
      c = __builtin_amdgcn_wmma_f32_16x16x32_bf16(false, a, false, bw1,
                                                  (short)0, c, false, false);
    }
    if (row0 + 16 <= n) {                       // full tile: straight stores
#pragma unroll
      for (int v = 0; v < 8; ++v)
        out[(long long)(row0 + v + 8 * hi) * DIM + col] = c[v] + bb;
    } else {                                    // ragged tail
#pragma unroll
      for (int v = 0; v < 8; ++v) {
        int r = row0 + v + 8 * hi;
        if (r < n) out[(long long)r * DIM + col] = c[v] + bb;
      }
    }
  }
}

// ---------------------------------------------------------------------------
// Edge kernel: persistent waves, one 16-edge tile per iteration:
//   1) E-tile = edge_attr_tile @ WE + bE via 8 WMMAs (WE fragments in registers)
//   2) stage 16x64 E tile in this wave's LDS slice; s_wait_dscnt 0 (wave-local)
//   3) per lane: one edge, 4 heads -> gather K[src],Q[dst],V[src] (L2), score,
//      exp(clip), scatter V*score with global_atomic_add_f32 (L2-resident out)
// ---------------------------------------------------------------------------
__global__ void __launch_bounds__(256)
edge_attn_kernel(const float* __restrict__ edge_attr,
                 const int*   __restrict__ edge_index,   // [2][NE]
                 const float* __restrict__ We, const float* __restrict__ be,
                 const float* __restrict__ Qd, const float* __restrict__ Kd,
                 const float* __restrict__ Vd,
                 float* __restrict__ out,
                 int ne, int ntiles) {
  __shared__ float Elds[8][16][ELDS_STRIDE];

  const int lane = threadIdx.x & 31;
  const int wave = threadIdx.x >> 5;   // 0..7
  const int nl   = lane & 15;
  const int hi   = lane >> 4;
  const float inv_sqrt_d = 0.35355339059327373f;   // 1/sqrt(8)

  // loop-invariant WE fragments (2 K-steps x 4 col tiles) + bias, in registers
  v16bf bw[2][4];
  float bias[4];
#pragma unroll
  for (int ks = 0; ks < 2; ++ks)
#pragma unroll
    for (int ct = 0; ct < 4; ++ct)
      bw[ks][ct] = load_b_bf16(We, DIM, ks * 32, ct * 16 + nl, lane);
#pragma unroll
  for (int ct = 0; ct < 4; ++ct) bias[ct] = be[ct * 16 + nl];

  const int tstride = gridDim.x * 8;
  for (int tile = blockIdx.x * 8 + wave; tile < ntiles; tile += tstride) {
    const int e0 = tile * 16;

    // prefetch this wave's next tile (gfx1250 global_prefetch_b8)
    {
      long long pf = ((long long)e0 + (long long)tstride * 16) * DIM;
      if (pf < (long long)ne * DIM) __builtin_prefetch(edge_attr + pf, 0, 1);
    }

    // --- 1) fused E projection: 8 x v_wmma_f32_16x16x32_bf16 ---
    v8f c[4] = {{}, {}, {}, {}};
#pragma unroll
    for (int ks = 0; ks < 2; ++ks) {
      v16bf a = load_a_bf16(edge_attr, DIM, e0, ne, ks * 32, lane);
#pragma unroll
      for (int ct = 0; ct < 4; ++ct)
        c[ct] = __builtin_amdgcn_wmma_f32_16x16x32_bf16(false, a, false, bw[ks][ct],
                                                        (short)0, c[ct],
                                                        false, false);
    }

    // --- 2) stage E tile in this wave's LDS slice ---
#pragma unroll
    for (int ct = 0; ct < 4; ++ct) {
      const int col = ct * 16 + nl;
#pragma unroll
      for (int v = 0; v < 8; ++v)
        Elds[wave][v + 8 * hi][col] = c[ct][v] + bias[ct];   // row M = v + 8*hi
    }
    // wave-local ordering: this wave reads only its own slice; LDS is in-order
    // per wave, the explicit split-counter wait + clobber pins the compiler.
    asm volatile("s_wait_dscnt 0x0" ::: "memory");

    // --- 3) score + scatter: lane = (edge nl, head half) ---
    const int eg = e0 + nl;
    if (eg < ne) {
      const int src = edge_index[eg];
      const int dst = edge_index[ne + eg];
      const float* __restrict__ Krow = Kd + (long long)src * DIM;
      const float* __restrict__ Qrow = Qd + (long long)dst * DIM;
      const float* __restrict__ Vrow = Vd + (long long)src * DIM;
      float*       __restrict__ orow = out + (long long)dst * DIM;

#pragma unroll
      for (int rep = 0; rep < 4; ++rep) {
        const int h  = hi * 4 + rep;      // lanes 0-15: heads 0-3; 16-31: 4-7
        const int cb = h * HDIM;
        float s = 0.0f;
#pragma unroll
        for (int d = 0; d < HDIM; ++d)
          s += Krow[cb + d] * Qrow[cb + d] * Elds[wave][nl][cb + d];
        s *= inv_sqrt_d;
        s = fminf(fmaxf(s, -5.0f), 5.0f);
        const float sc = __expf(s);
#pragma unroll
        for (int d = 0; d < HDIM; ++d)
          unsafeAtomicAdd(&orow[cb + d], Vrow[cb + d] * sc);  // global_atomic_add_f32
      }
    }
  }
}

// ---------------------------------------------------------------------------
extern "C" void kernel_launch(void* const* d_in, const int* in_sizes, int n_in,
                              void* d_out, int out_size, void* d_ws, size_t ws_size,
                              hipStream_t stream) {
  const float* x         = (const float*)d_in[0];
  const float* edge_attr = (const float*)d_in[1];
  const int*   edge_idx  = (const int*)  d_in[2];
  const float* WQ = (const float*)d_in[3];
  const float* bQ = (const float*)d_in[4];
  const float* WK = (const float*)d_in[5];
  const float* bK = (const float*)d_in[6];
  const float* WE = (const float*)d_in[7];
  const float* bE = (const float*)d_in[8];
  const float* WV = (const float*)d_in[9];
  const float* bV = (const float*)d_in[10];
  float* out = (float*)d_out;

  const int n  = in_sizes[0] / DIM;
  const int ne = in_sizes[1] / DIM;

  float* Qd = (float*)d_ws;                 // N*64 f32
  float* Kd = Qd + (long long)n * DIM;      // N*64 f32
  float* Vd = Kd + (long long)n * DIM;      // N*64 f32

  // 1) zero the scatter-add target (d_out is poisoned by the harness).
  //    N*64 is a multiple of 4 -> vectorized b128 stores.
  const long long n4 = (long long)n * DIM / 4;
  zero_f32_kernel<<<(int)((n4 + 255) / 256), 256, 0, stream>>>(out, n4);

  // 2) node projections Q/K/V (persistent waves, WMMA bf16 / f32 accumulate)
  const int ntiles_n = (n + 15) / 16;
  int grid_n = ntiles_n < 1536 ? ntiles_n : 1536;
  node_proj_kernel<<<grid_n, 384, 0, stream>>>(
      x, WQ, bQ, WK, bK, WV, bV, Qd, Kd, Vd, n, ntiles_n);

  // 3) fused E-projection + attention + scatter (persistent waves)
  const int ntiles_e = (ne + 15) / 16;
  int grid_e = (ntiles_e + 7) / 8;
  if (grid_e > 2048) grid_e = 2048;
  edge_attn_kernel<<<grid_e, 256, 0, stream>>>(
      edge_attr, edge_idx, WE, bE, Qd, Kd, Vd, out, ne, ntiles_e);
}
